// HyperLSTMCell_12378095747447
// MI455X (gfx1250) — compile-verified, matrix operand order checked by reading, not run on hardware
//
#include <hip/hip_runtime.h>
#include <hip/hip_bf16.h>

#define B_SZ   2048
#define IN_SZ  1024
#define N_SZ   1024
#define HY_SZ  256
#define E_SZ   32
#define EPS    1e-5f
#define FBIAS  1.0f

typedef __attribute__((ext_vector_type(16))) __bf16 bf16x16;
typedef __attribute__((ext_vector_type(2)))  __bf16 bf16x2;
typedef __attribute__((ext_vector_type(8)))  float  f32x8;
typedef __attribute__((ext_vector_type(8)))  unsigned int u32x8;

struct Seg {
    const float* A;   // [M,K] row-major
    int lda;
    const float* Bm;  // [K,N] row-major
    int ldb;
    int K;            // multiple of 32
};

// hardware bf16 convert (RNE) -> packed dword (v_cvt_pk_bf16_f32)
__device__ __forceinline__ unsigned int pack2bf(float lo, float hi) {
    bf16x2 p = { (__bf16)lo, (__bf16)hi };
    return __builtin_bit_cast(unsigned int, p);
}
__device__ __forceinline__ float sigmoidf(float x) { return 1.0f / (1.0f + __expf(-x)); }

__device__ __forceinline__ float block_sum256(float v, float* red) {
    int t = threadIdx.x;
    red[t] = v; __syncthreads();
    #pragma unroll
    for (int s = 128; s > 0; s >>= 1) {
        if (t < s) red[t] += red[t + s];
        __syncthreads();
    }
    float r = red[0]; __syncthreads();
    return r;
}

// stage one 128x32 A tile (row-major) and one 128(n)x32(k) transposed B tile.
// Phase 1 issues ALL global loads back-to-back (pipelined latency),
// phase 2 converts to bf16 and writes LDS.
__device__ __forceinline__ void stage_tiles(unsigned short (*Asb)[48],
                                            unsigned short (*Bsb)[48],
                                            const Seg& sg, int m0, int n0,
                                            int k0, int t) {
    float4 av[4];
    float  bv[4][4];
    #pragma unroll
    for (int i = 0; i < 4; ++i) {
        int q   = i * 256 + t;        // 0..1023 quads
        int row = q >> 3;
        int k4  = (q & 7) * 4;
        av[i] = *(const float4*)(sg.A + (size_t)(m0 + row) * sg.lda + (k0 + k4));
    }
    #pragma unroll
    for (int i = 0; i < 4; ++i) {
        int q  = i * 256 + t;
        int n  = q & 127;
        int k4 = (q >> 7) * 4;
        const float* bp = sg.Bm + (size_t)(k0 + k4) * sg.ldb + (n0 + n);
        bv[i][0] = bp[0];
        bv[i][1] = bp[(size_t)sg.ldb];
        bv[i][2] = bp[2 * (size_t)sg.ldb];
        bv[i][3] = bp[3 * (size_t)sg.ldb];
    }
    #pragma unroll
    for (int i = 0; i < 4; ++i) {
        int q   = i * 256 + t;
        int row = q >> 3;
        int k4  = (q & 7) * 4;
        *(uint2*)&Asb[row][k4] =
            make_uint2(pack2bf(av[i].x, av[i].y), pack2bf(av[i].z, av[i].w));
    }
    #pragma unroll
    for (int i = 0; i < 4; ++i) {
        int q  = i * 256 + t;
        int n  = q & 127;
        int k4 = (q >> 7) * 4;
        *(uint2*)&Bsb[n][k4] =
            make_uint2(pack2bf(bv[i][0], bv[i][1]), pack2bf(bv[i][2], bv[i][3]));
    }
}

// ---------------------------------------------------------------------------
// Multi-segment GEMM: C[M,N] = sum_seg A_seg[M,K_seg] * B_seg[K_seg,N]
// fp32 in/out, bf16 WMMA math, fp32 register accumulation across segments,
// double-buffered LDS (one barrier per k-step), C written exactly once.
// Block: 256 threads = 8 waves; C tile 128x128; each wave 32x64 (2x4 WMMA).
// ---------------------------------------------------------------------------
template <int NSEG>
__global__ __launch_bounds__(256)
void wmma_gemm_multi(Seg s0, Seg s1, Seg s2, float* __restrict__ C, int ldc) {
    __shared__ unsigned short As[2][128][48];   // 96B rows -> 16B-aligned b128
    __shared__ unsigned short Bs[2][128][48];

    const int t     = threadIdx.x;
    const int wid   = t >> 5;
    const int lane  = t & 31;
    const int wm    = wid & 3;       // 0..3 -> 32-row strip
    const int wn    = wid >> 2;      // 0..1 -> 64-col strip
    const int m0    = blockIdx.y * 128;
    const int n0    = blockIdx.x * 128;
    const int khalf = lane >> 4;
    const int mn    = lane & 15;

    f32x8 acc[2][4];
    #pragma unroll
    for (int i = 0; i < 2; ++i)
        #pragma unroll
        for (int j = 0; j < 4; ++j) acc[i][j] = 0.f;

    const Seg segs[3] = {s0, s1, s2};
    #pragma unroll
    for (int sgi = 0; sgi < NSEG; ++sgi) {
        const Seg sg = segs[sgi];
        const int nk = sg.K >> 5;

        stage_tiles(As[0], Bs[0], sg, m0, n0, 0, t);
        __syncthreads();

        for (int ki = 0; ki < nk; ++ki) {
            const int cur = ki & 1;
            if (ki + 1 < nk)                       // stage next slice into the
                stage_tiles(As[cur ^ 1], Bs[cur ^ 1], sg, m0, n0,
                            (ki + 1) << 5, t);     // other buffer (no hazard)

            // --- fragments (16-bit 16x32 A layout: K pairs per VGPR)
            bf16x16 af[2], bfr[4];
            #pragma unroll
            for (int s = 0; s < 2; ++s) {
                int m = wm * 32 + s * 16 + mn;
                u32x8 u;
                #pragma unroll
                for (int v = 0; v < 8; ++v) {
                    int kk = ((v < 4) ? (2 * v) : (16 + 2 * (v - 4))) + khalf * 8;
                    u[v] = *(const unsigned int*)&As[cur][m][kk];
                }
                af[s] = __builtin_bit_cast(bf16x16, u);
            }
            #pragma unroll
            for (int s = 0; s < 4; ++s) {
                int n = wn * 64 + s * 16 + mn;
                u32x8 u;
                #pragma unroll
                for (int v = 0; v < 8; ++v) {
                    int kk = ((v < 4) ? (2 * v) : (16 + 2 * (v - 4))) + khalf * 8;
                    u[v] = *(const unsigned int*)&Bs[cur][n][kk];
                }
                bfr[s] = __builtin_bit_cast(bf16x16, u);
            }
            #pragma unroll
            for (int si = 0; si < 2; ++si)
                #pragma unroll
                for (int sj = 0; sj < 4; ++sj)
                    acc[si][sj] = __builtin_amdgcn_wmma_f32_16x16x32_bf16(
                        false, af[si], false, bfr[sj], (short)0, acc[si][sj],
                        false, false);
            __syncthreads();
        }
    }

    // --- epilogue: f32 16x16 C layout (VGPR r: lanes 0-15 M=r, 16-31 M=r+8)
    const int mlo = (lane >> 4) * 8;
    const int nlo = lane & 15;
    #pragma unroll
    for (int si = 0; si < 2; ++si) {
        #pragma unroll
        for (int sj = 0; sj < 4; ++sj) {
            int mb = m0 + wm * 32 + si * 16 + mlo;
            int nb = n0 + wn * 64 + sj * 16 + nlo;
            #pragma unroll
            for (int r = 0; r < 8; ++r)
                C[(size_t)(mb + r) * ldc + nb] = acc[si][sj][r];
        }
    }
}

// ---------------------------------------------------------------------------
// Inner (hyper) LSTM pointwise: LN over 1024, gates, LN over 256.
// ---------------------------------------------------------------------------
__global__ __launch_bounds__(256)
void hyper_pointwise(const float* __restrict__ ifgo, const float* __restrict__ c_hat,
                     const float* __restrict__ hg, const float* __restrict__ hb,
                     const float* __restrict__ gc, const float* __restrict__ bc,
                     float* __restrict__ ho_ws, float* __restrict__ hhat_out,
                     float* __restrict__ chat_out) {
    __shared__ float xs[1024];
    __shared__ float red[256];
    const int b = blockIdx.x;
    const int t = threadIdx.x;

    float lsum = 0.f;
    #pragma unroll
    for (int i = 0; i < 4; ++i) {
        int idx = t + i * 256;
        float v = ifgo[(size_t)b * 1024 + idx];
        xs[idx] = v;
        lsum += v;
    }
    float mean = block_sum256(lsum, red) * (1.0f / 1024.0f);
    float vsum = 0.f;
    #pragma unroll
    for (int i = 0; i < 4; ++i) {
        float d = xs[t + i * 256] - mean;
        vsum += d * d;
    }
    float var  = block_sum256(vsum, red) * (1.0f / 1024.0f);
    float rstd = rsqrtf(var + EPS);

    float i_ = (xs[t]       - mean) * rstd * hg[t]       + hb[t];
    float j_ = (xs[256 + t] - mean) * rstd * hg[256 + t] + hb[256 + t];
    float f_ = (xs[512 + t] - mean) * rstd * hg[512 + t] + hb[512 + t];
    float o_ = (xs[768 + t] - mean) * rstd * hg[768 + t] + hb[768 + t];

    float cn = c_hat[(size_t)b * 256 + t] * sigmoidf(f_ + FBIAS)
             + sigmoidf(i_) * tanhf(j_);
    chat_out[(size_t)b * 256 + t] = cn;

    float cmean = block_sum256(cn, red) * (1.0f / 256.0f);
    float cd    = cn - cmean;
    float cvar  = block_sum256(cd * cd, red) * (1.0f / 256.0f);
    float crstd = rsqrtf(cvar + EPS);
    float h = sigmoidf(o_) * tanhf((cn - cmean) * crstd * gc[t] + bc[t]);
    hhat_out[(size_t)b * 256 + t] = h;
    ho_ws[(size_t)b * 256 + t]   = h;
}

// ---------------------------------------------------------------------------
// z = ho @ [zw_x_w | zw_h_w | zb_h_w] (+ biases) : per row, 384 outputs.
// ---------------------------------------------------------------------------
__global__ __launch_bounds__(384)
void z_kernel(const float* __restrict__ ho,
              const float* __restrict__ zw_x_w, const float* __restrict__ zw_x_b,
              const float* __restrict__ zw_h_w, const float* __restrict__ zw_h_b,
              const float* __restrict__ zb_h_w,
              float* __restrict__ z) {
    __shared__ float hs[256];
    const int b = blockIdx.x;
    const int t = threadIdx.x;
    if (t < 256) hs[t] = ho[(size_t)b * 256 + t];
    __syncthreads();

    const int group = t >> 7;          // 0: zx, 1: zh, 2: zb
    const int r = t & 127;
    const int g = r >> 5;
    const int e = r & 31;
    const float* W = (group == 0) ? zw_x_w : (group == 1) ? zw_h_w : zb_h_w;
    float s = (group == 0) ? zw_x_b[g * 32 + e]
            : (group == 1) ? zw_h_b[g * 32 + e] : 0.f;
    #pragma unroll 8
    for (int hy = 0; hy < 256; ++hy)
        s += hs[hy] * W[(size_t)(g * 256 + hy) * 32 + e];
    z[(size_t)b * 384 + t] = s;
}

// ---------------------------------------------------------------------------
// Main LSTM pointwise: hypernorm expand (E=32), LN 4096, gates, LN 1024.
// ---------------------------------------------------------------------------
__global__ __launch_bounds__(256)
void main_pointwise(const float* __restrict__ xh, const float* __restrict__ hh,
                    const float* __restrict__ z,
                    const float* __restrict__ alpha_x, const float* __restrict__ alpha_h,
                    const float* __restrict__ beta_h,  const float* __restrict__ bias,
                    const float* __restrict__ lg, const float* __restrict__ lb,
                    const float* __restrict__ gc, const float* __restrict__ bc,
                    const float* __restrict__ c_in,
                    float* __restrict__ h_out, float* __restrict__ c_out) {
    __shared__ float zs[384];
    __shared__ float gs[4096];
    __shared__ float cs[1024];
    __shared__ float red[256];
    const int b = blockIdx.x;
    const int t = threadIdx.x;

    if (t < 256) {
        zs[t] = z[(size_t)b * 384 + t];
        if (t < 128) zs[256 + t] = z[(size_t)b * 384 + 256 + t];
    }
    __syncthreads();

    float lsum = 0.f;
    #pragma unroll
    for (int i = 0; i < 16; ++i) {
        int idx = t + i * 256;
        int g = idx >> 10, n = idx & 1023;
        const float* zx = zs + g * 32;
        const float* zh = zs + 128 + g * 32;
        const float* zb = zs + 256 + g * 32;
        float ax = 0.f, ah = 0.f, bh = 0.f;
        #pragma unroll 8
        for (int e = 0; e < 32; ++e) {
            size_t wo = (size_t)(g * 32 + e) * 1024 + n;
            ax += zx[e] * alpha_x[wo];
            ah += zh[e] * alpha_h[wo];
            bh += zb[e] * beta_h[wo];
        }
        float v = ax * xh[(size_t)b * 4096 + idx]
                + ah * hh[(size_t)b * 4096 + idx]
                + bh + bias[idx];
        gs[idx] = v;
        lsum += v;
    }
    float mean = block_sum256(lsum, red) * (1.0f / 4096.0f);
    float vsum = 0.f;
    #pragma unroll
    for (int i = 0; i < 16; ++i) {
        float d = gs[t + i * 256] - mean;
        vsum += d * d;
    }
    float var  = block_sum256(vsum, red) * (1.0f / 4096.0f);
    float rstd = rsqrtf(var + EPS);

    float o_loc[4], c_loc[4];
    float csum = 0.f;
    #pragma unroll
    for (int i = 0; i < 4; ++i) {
        int n = t + i * 256;
        float i_ = (gs[n]        - mean) * rstd * lg[n]        + lb[n];
        float j_ = (gs[1024 + n] - mean) * rstd * lg[1024 + n] + lb[1024 + n];
        float f_ = (gs[2048 + n] - mean) * rstd * lg[2048 + n] + lb[2048 + n];
        float o_ = (gs[3072 + n] - mean) * rstd * lg[3072 + n] + lb[3072 + n];
        float cn = c_in[(size_t)b * 1024 + n] * sigmoidf(f_ + FBIAS)
                 + sigmoidf(i_) * tanhf(j_);
        c_out[(size_t)b * 1024 + n] = cn;
        cs[n] = cn;
        o_loc[i] = o_;
        c_loc[i] = cn;
        csum += cn;
    }
    float cmean = block_sum256(csum, red) * (1.0f / 1024.0f);
    float cvsum = 0.f;
    #pragma unroll
    for (int i = 0; i < 4; ++i) {
        float d = cs[t + i * 256] - cmean;
        cvsum += d * d;
    }
    float cvar  = block_sum256(cvsum, red) * (1.0f / 1024.0f);
    float crstd = rsqrtf(cvar + EPS);
    #pragma unroll
    for (int i = 0; i < 4; ++i) {
        int n = t + i * 256;
        float ln = (c_loc[i] - cmean) * crstd * gc[n] + bc[n];
        h_out[(size_t)b * 1024 + n] = sigmoidf(o_loc[i]) * tanhf(ln);
    }
}

// ---------------------------------------------------------------------------
extern "C" void kernel_launch(void* const* d_in, const int* in_sizes, int n_in,
                              void* d_out, int out_size, void* d_ws, size_t ws_size,
                              hipStream_t stream) {
    (void)in_sizes; (void)n_in; (void)out_size; (void)ws_size;
    const float* x        = (const float*)d_in[0];
    const float* h        = (const float*)d_in[1];
    const float* c        = (const float*)d_in[2];
    const float* h_hat    = (const float*)d_in[3];
    const float* c_hat    = (const float*)d_in[4];
    const float* hyper_Wi = (const float*)d_in[5];
    const float* hyper_Wh = (const float*)d_in[6];
    const float* h_lag    = (const float*)d_in[7];
    const float* h_lab    = (const float*)d_in[8];
    const float* h_lcg    = (const float*)d_in[9];
    const float* h_lcb    = (const float*)d_in[10];
    const float* w_x      = (const float*)d_in[11];
    const float* w_h      = (const float*)d_in[12];
    const float* zw_x_w   = (const float*)d_in[13];
    const float* zw_x_b   = (const float*)d_in[14];
    const float* alpha_x  = (const float*)d_in[15];
    const float* zw_h_w   = (const float*)d_in[16];
    const float* zw_h_b   = (const float*)d_in[17];
    const float* alpha_h  = (const float*)d_in[18];
    const float* zb_h_w   = (const float*)d_in[19];
    const float* beta_h   = (const float*)d_in[20];
    const float* bias     = (const float*)d_in[21];
    const float* ln_all_g = (const float*)d_in[22];
    const float* ln_all_b = (const float*)d_in[23];
    const float* ln_c_g   = (const float*)d_in[24];
    const float* ln_c_b   = (const float*)d_in[25];

    float* ws   = (float*)d_ws;
    float* ifgo = ws;                       // 2048*1024
    float* xh   = ws + 2097152;             // 2048*4096
    float* hh   = ws + 10485760;            // 2048*4096
    float* ho   = ws + 18874368;            // 2048*256
    float* zz   = ws + 19398656;            // 2048*384

    float* h_next = (float*)d_out;
    float* c_next = h_next + (size_t)B_SZ * N_SZ;
    float* hhat_n = c_next + (size_t)B_SZ * N_SZ;
    float* chat_n = hhat_n + (size_t)B_SZ * HY_SZ;

    dim3 blk(256);
    Seg segX  = {x,     IN_SZ, hyper_Wi,                          1024, 1024};
    Seg segH  = {h,     N_SZ,  hyper_Wi + (size_t)1024 * 1024,    1024, 1024};
    Seg segHH = {h_hat, HY_SZ, hyper_Wh,                          1024, 256};
    Seg segWX = {x,     IN_SZ, w_x,                               4096, 1024};
    Seg segWH = {h,     N_SZ,  w_h,                               4096, 1024};
    Seg segZ  = {nullptr, 0, nullptr, 0, 0};

    // inner-cell pre-activations: ifgo = x@Wi[:1024] + h@Wi[1024:] + h_hat@Wh
    wmma_gemm_multi<3><<<dim3(1024 / 128, 2048 / 128), blk, 0, stream>>>(
        segX, segH, segHH, ifgo, 1024);

    hyper_pointwise<<<B_SZ, 256, 0, stream>>>(
        ifgo, c_hat, h_lag, h_lab, h_lcg, h_lcb, ho, hhat_n, chat_n);

    // main-cell GEMMs
    wmma_gemm_multi<1><<<dim3(4096 / 128, 2048 / 128), blk, 0, stream>>>(
        segWX, segZ, segZ, xh, 4096);
    wmma_gemm_multi<1><<<dim3(4096 / 128, 2048 / 128), blk, 0, stream>>>(
        segWH, segZ, segZ, hh, 4096);

    z_kernel<<<B_SZ, 384, 0, stream>>>(ho, zw_x_w, zw_x_b, zw_h_w, zw_h_b, zb_h_w, zz);

    main_pointwise<<<B_SZ, 256, 0, stream>>>(
        xh, hh, zz, alpha_x, alpha_h, beta_h, bias,
        ln_all_g, ln_all_b, ln_c_g, ln_c_b, c, h_next, c_next);
}
